// ColorNet_58961311039891
// MI455X (gfx1250) — matrix operand order repaired and checked by reading, compiled, not run on hardware
//
#include <hip/hip_runtime.h>

typedef __attribute__((ext_vector_type(16))) __bf16 v16bf;
typedef __attribute__((ext_vector_type(8)))  __bf16 v8bf;
typedef __attribute__((ext_vector_type(8)))  float  v8f;

#define HW   50176
#define IMGW 224

// ---------------------------------------------------------------------------
// Implicit-GEMM conv (layers 1..13): act bf16 [p][cin] -> out f32 [p][cout]
// Block: 256 thr = 8 waves. Tile: 64 cout x 128 pixels. Wave: 16 x 64 (4 accs).
// Border handling via zero-row base-pointer select (no predication in hot loop).
// ---------------------------------------------------------------------------
__global__ __launch_bounds__(256) void conv_gemm_bf16(
    const __bf16* __restrict__ act, const __bf16* __restrict__ wtap,
    const float* __restrict__ bias, const __bf16* __restrict__ zrow,
    float* __restrict__ out, int cin, int cout, int ksize)
{
  const int p0   = blockIdx.x * 128;       // pixel tile
  const int m0   = blockIdx.y * 64;        // cout tile
  const int lane = threadIdx.x & 31;
  const int wv   = threadIdx.x >> 5;       // wave 0..7
  const int mw   = (wv >> 1) * 16;         // wave M offset: 0,16,32,48
  const int nw   = (wv & 1) * 64;          // wave N offset: 0,64
  const int hl   = lane >> 4;              // lane half
  const int lid  = lane & 15;

  v8f acc[4] = {{}, {}, {}, {}};

  // Pixels owned by this lane's four B fragments (C-fragment columns too).
  int pB[4], hB[4], wB[4];
#pragma unroll
  for (int j = 0; j < 4; ++j) {
    pB[j] = p0 + nw + j * 16 + lid;
    hB[j] = pB[j] / IMGW;
    wB[j] = pB[j] - hB[j] * IMGW;
  }

  const int    taps = ksize * ksize;
  const size_t cc   = (size_t)cout * cin;
  // A-fragment row: both lane halves read row M=lid; K runs offset by hl*16.
  const __bf16* wrowBase = wtap + (size_t)(m0 + mw + lid) * cin + hl * 16;

  for (int tap = 0; tap < taps; ++tap) {
    const int dy = (ksize == 3) ? (tap / 3 - 1) : 0;
    const int dx = (ksize == 3) ? (tap - (tap / 3) * 3 - 1) : 0;

    // Per-tap base pointers: shifted activation row, or the zero row when the
    // shifted pixel is out of bounds. Loads in the K loop are unconditional.
    const __bf16* bptr[4];
#pragma unroll
    for (int j = 0; j < 4; ++j) {
      const bool v = ((unsigned)(hB[j] + dy) < (unsigned)IMGW) &&
                     ((unsigned)(wB[j] + dx) < (unsigned)IMGW);
      const long sp = (long)(pB[j] + dy * IMGW + dx) * cin;
      bptr[j] = (v ? act + sp : zrow) + hl * 16;
    }
    const __bf16* arow = wrowBase + (size_t)tap * cc;

    for (int c0 = 0; c0 < cin; c0 += 32) {
      // A: lanes 0-15 hold K 0-15, lanes 16-31 hold K 16-31 (matches B);
      // two contiguous 16B chunks per lane.
      v8bf alo = *(const v8bf*)(arow + c0);
      v8bf ahi = *(const v8bf*)(arow + c0 + 8);
      v16bf afrag;
#pragma unroll
      for (int i = 0; i < 8; ++i) { afrag[i] = alo[i]; afrag[8 + i] = ahi[i]; }

      // B: lanes 0-15 = K 0-15 at pixel n, lanes 16-31 = K 16-31; with
      // channel-contiguous activations this is one aligned 32B load per lane.
#pragma unroll
      for (int j = 0; j < 4; ++j) {
        v16bf bf = *(const v16bf*)(bptr[j] + c0);
        acc[j] = __builtin_amdgcn_wmma_f32_16x16x32_bf16(false, afrag, false, bf,
                                                         (short)0, acc[j], false, false);
      }
    }
  }

  // C layout: VGPR r, lane half h -> channel m0+mw+8h+r; 8 consecutive
  // channels per lane -> contiguous f32 stores into [pixel][channel].
  const int cb = m0 + mw + hl * 8;
#pragma unroll
  for (int j = 0; j < 4; ++j) {
    float* d = out + (size_t)pB[j] * cout + cb;
#pragma unroll
    for (int r = 0; r < 8; ++r) d[r] = acc[j][r] + bias[cb + r];
  }
}

// ---------------------------------------------------------------------------
// Zero the border zero-row (ws is poisoned by the harness; re-zero per launch)
// ---------------------------------------------------------------------------
__global__ void zero_row(__bf16* __restrict__ z)
{
  z[blockIdx.x * 256 + threadIdx.x] = (__bf16)0.0f;
}

// ---------------------------------------------------------------------------
// Weight repack: f32 (cout,cin,kh,kw) -> bf16 tap-major [tap][cout][cin]
// ---------------------------------------------------------------------------
__global__ void convert_w(const float* __restrict__ w, __bf16* __restrict__ wtap,
                          int cout, int cin, int khw)
{
  size_t idx   = (size_t)blockIdx.x * blockDim.x + threadIdx.x;
  size_t total = (size_t)cout * cin * khw;
  if (idx >= total) return;
  size_t cc  = (size_t)cout * cin;
  int    tap = (int)(idx / cc);
  size_t rem = idx - (size_t)tap * cc;
  int    co  = (int)(rem / cin);
  int    ci  = (int)(rem - (size_t)co * cin);
  wtap[idx] = (__bf16)w[((size_t)co * cin + ci) * khw + tap];
}

// ---------------------------------------------------------------------------
// Layer 0: 1->64 3x3 direct (K=9, too small for WMMA; <0.01% of total FLOPs)
// ---------------------------------------------------------------------------
__global__ void conv0_direct(const float* __restrict__ x, const float* __restrict__ w,
                             const float* __restrict__ b, float* __restrict__ out)
{
  int c = threadIdx.x & 63;
  int p = blockIdx.x * 4 + (threadIdx.x >> 6);
  int h = p / IMGW, wd = p - h * IMGW;
  float s = b[c];
#pragma unroll
  for (int ky = 0; ky < 3; ++ky)
#pragma unroll
    for (int kx = 0; kx < 3; ++kx) {
      int hh = h + ky - 1, ww = wd + kx - 1;
      if ((unsigned)hh < (unsigned)IMGW && (unsigned)ww < (unsigned)IMGW)
        s += x[hh * IMGW + ww] * w[c * 9 + ky * 3 + kx];
    }
  out[(size_t)p * 64 + c] = s;
}

// ---------------------------------------------------------------------------
// Layer 14: 64->3 1x1 direct
// ---------------------------------------------------------------------------
__global__ void conv14_direct(const __bf16* __restrict__ act, const float* __restrict__ w,
                              const float* __restrict__ b, float* __restrict__ out)
{
  int idx = blockIdx.x * blockDim.x + threadIdx.x;
  if (idx >= HW * 3) return;
  int p = idx / 3, co = idx - p * 3;
  float s = b[co];
  const __bf16* arow = act + (size_t)p * 64;
  const float*  wrow = w + co * 64;
#pragma unroll
  for (int ci = 0; ci < 64; ++ci) s += (float)arow[ci] * wrow[ci];
  out[(size_t)p * 3 + co] = s;
}

// ---------------------------------------------------------------------------
// BN stats, deterministic two-stage (no float atomics). buf is f32 [p][C].
// Stage 1: block b reduces one image row (224 pixels) for all channels.
// ---------------------------------------------------------------------------
__global__ void stats_partial(const float* __restrict__ buf, float* __restrict__ part, int C)
{
  int b = blockIdx.x;  // 0..223
  for (int c = threadIdx.x; c < C; c += blockDim.x) {
    float s = 0.f, q = 0.f;
    const float* p0 = buf + (size_t)b * 224 * C + c;
    for (int i = 0; i < 224; ++i) { float v = p0[(size_t)i * C]; s += v; q += v * v; }
    part[(size_t)b * 2 * C + c]     = s;
    part[(size_t)b * 2 * C + C + c] = q;
  }
}

__global__ void stats_final(const float* __restrict__ part, float* __restrict__ mean,
                            float* __restrict__ rstd, int C)
{
  int c = blockIdx.x * blockDim.x + threadIdx.x;
  if (c >= C) return;
  float s = 0.f, q = 0.f;
  for (int b = 0; b < 224; ++b) {
    s += part[(size_t)b * 2 * C + c];
    q += part[(size_t)b * 2 * C + C + c];
  }
  float m = s * (1.f / (float)HW);
  float v = q * (1.f / (float)HW) - m * m;
  mean[c] = m;
  rstd[c] = rsqrtf(fmaxf(v, 0.f) + 1e-5f);
}

// Normalize + scale/shift + ReLU + cast to bf16 act [p][C] (C power of two)
__global__ void bn_relu_bf16(const float* __restrict__ buf, const float* __restrict__ mean,
                             const float* __restrict__ rstd, const float* __restrict__ gamma,
                             const float* __restrict__ beta, __bf16* __restrict__ outa,
                             int cmask, long total)
{
  long idx = (long)blockIdx.x * blockDim.x + threadIdx.x;
  if (idx >= total) return;
  int c = (int)idx & cmask;
  float v = (buf[idx] - mean[c]) * rstd[c] * gamma[c] + beta[c];
  outa[idx] = (__bf16)fmaxf(v, 0.f);
}

// Final layer: normalize C=3, write f32 NCHW x into d_out
__global__ void bn_relu_final(const float* __restrict__ buf, const float* __restrict__ mean,
                              const float* __restrict__ rstd, const float* __restrict__ gamma,
                              const float* __restrict__ beta, float* __restrict__ xout)
{
  int p = blockIdx.x * blockDim.x + threadIdx.x;
  if (p >= HW) return;
#pragma unroll
  for (int c = 0; c < 3; ++c) {
    float v = (buf[(size_t)p * 3 + c] - mean[c]) * rstd[c] * gamma[c] + beta[c];
    xout[(size_t)c * HW + p] = fmaxf(v, 0.f);
  }
}

__global__ void luminance(const float* __restrict__ x, float* __restrict__ y)
{
  int p = blockIdx.x * blockDim.x + threadIdx.x;
  if (p >= HW) return;
  y[p] = 0.2125f * x[p] + 0.7154f * x[HW + p] + 0.0721f * x[2 * HW + p];
}

// ---------------------------------------------------------------------------
extern "C" void kernel_launch(void* const* d_in, const int* in_sizes, int n_in,
                              void* d_out, int out_size, void* d_ws, size_t ws_size,
                              hipStream_t stream)
{
  struct LC { int cin, cout, k; };
  static const LC L[15] = {{1,64,3},{64,128,1},{128,128,3},{128,256,1},{256,256,3},
                           {256,256,1},{256,256,3},{256,512,1},{512,512,3},{512,512,1},
                           {512,512,3},{512,256,1},{256,128,1},{128,64,1},{64,3,1}};
  (void)in_sizes; (void)n_in; (void)out_size; (void)ws_size;

  const float* x = (const float*)d_in[0];
  auto Wp = [&](int l) { return (const float*)d_in[2 + 4 * l + 0]; };
  auto Bp = [&](int l) { return (const float*)d_in[2 + 4 * l + 1]; };
  auto Gp = [&](int l) { return (const float*)d_in[2 + 4 * l + 2]; };
  auto Ep = [&](int l) { return (const float*)d_in[2 + 4 * l + 3]; };

  // Workspace carve-up (~220 MB): f32 conv buffer + bf16 act ping-pong +
  // stats partials + zero row + bf16 tap-major weights.
  char*  ws  = (char*)d_ws;
  size_t off = 0;
  float*  convbuf = (float*)(ws + off);  off += (size_t)HW * 512 * sizeof(float);
  __bf16* actA    = (__bf16*)(ws + off); off += (size_t)HW * 512 * sizeof(__bf16);
  __bf16* actB    = (__bf16*)(ws + off); off += (size_t)HW * 512 * sizeof(__bf16);
  float*  part    = (float*)(ws + off);  off += (size_t)224 * 2 * 512 * sizeof(float);
  float*  meanb   = (float*)(ws + off);  off += 512 * sizeof(float);
  float*  rstdb   = (float*)(ws + off);  off += 512 * sizeof(float);
  __bf16* zrow    = (__bf16*)(ws + off); off += 512 * sizeof(__bf16);
  off = (off + 255) & ~(size_t)255;
  __bf16* wtap[15];
  for (int l = 1; l <= 13; ++l) {
    wtap[l] = (__bf16*)(ws + off);
    off += (size_t)L[l].cin * L[l].cout * L[l].k * L[l].k * sizeof(__bf16);
    off = (off + 255) & ~(size_t)255;
  }

  // Per-launch init: zero the border row, repack weights to bf16 tap-major.
  zero_row<<<dim3(2), 256, 0, stream>>>(zrow);
  for (int l = 1; l <= 13; ++l) {
    size_t total = (size_t)L[l].cin * L[l].cout * L[l].k * L[l].k;
    convert_w<<<dim3((unsigned)((total + 255) / 256)), 256, 0, stream>>>(
        Wp(l), wtap[l], L[l].cout, L[l].cin, L[l].k * L[l].k);
  }

  // Layer 0 (direct) + BN/ReLU -> actA
  conv0_direct<<<dim3(HW / 4), 256, 0, stream>>>(x, Wp(0), Bp(0), convbuf);
  stats_partial<<<224, 256, 0, stream>>>(convbuf, part, 64);
  stats_final<<<1, 256, 0, stream>>>(part, meanb, rstdb, 64);
  bn_relu_bf16<<<dim3((unsigned)(((long)HW * 64 + 255) / 256)), 256, 0, stream>>>(
      convbuf, meanb, rstdb, Gp(0), Ep(0), actA, 63, (long)HW * 64);

  // Layers 1..13: WMMA implicit-GEMM conv + BN/ReLU, ping-pong activations.
  __bf16* ain = actA;
  __bf16* aout = actB;
  for (int l = 1; l <= 13; ++l) {
    int cin = L[l].cin, cout = L[l].cout, k = L[l].k;
    conv_gemm_bf16<<<dim3(HW / 128, cout / 64), 256, 0, stream>>>(
        ain, wtap[l], Bp(l), zrow, convbuf, cin, cout, k);
    stats_partial<<<224, 256, 0, stream>>>(convbuf, part, cout);
    stats_final<<<dim3((cout + 255) / 256), 256, 0, stream>>>(part, meanb, rstdb, cout);
    bn_relu_bf16<<<dim3((unsigned)(((long)HW * cout + 255) / 256)), 256, 0, stream>>>(
        convbuf, meanb, rstdb, Gp(l), Ep(l), aout, cout - 1, (long)HW * cout);
    __bf16* t = ain; ain = aout; aout = t;
  }

  // Layer 14 (direct, cout=3) + BN/ReLU -> d_out x, then luminance -> d_out y.
  conv14_direct<<<dim3((HW * 3 + 255) / 256), 256, 0, stream>>>(ain, Wp(14), Bp(14), convbuf);
  stats_partial<<<224, 256, 0, stream>>>(convbuf, part, 3);
  stats_final<<<1, 256, 0, stream>>>(part, meanb, rstdb, 3);
  float* xout = (float*)d_out;
  bn_relu_final<<<dim3((HW + 255) / 256), 256, 0, stream>>>(convbuf, meanb, rstdb,
                                                            Gp(14), Ep(14), xout);
  luminance<<<dim3((HW + 255) / 256), 256, 0, stream>>>(xout, xout + 3 * HW);
}
